// GPTNeoSelfAttention_6820408066119
// MI455X (gfx1250) — matrix operand order repaired and checked by reading, not compiled, not run on hardware
//
#include <hip/hip_runtime.h>

// ---------------------------------------------------------------------------
// GPT-Neo self-attention (B=1, S=4096, E=1024, H=16, D=64),
// heads 0-7 causal, heads 8-15 sliding window 256.
// bf16 WMMA (v_wmma_f32_16x16x32_bf16), f32 accumulate.
//  - x and weights pre-converted to bf16 (weights also transposed)
//  - GEMM: 32x512 block tile, 8 wmma per wave per k-step, padded LDS rows
//  - double-buffered async global->LDS staging (ASYNCcnt), SADDR form
//  - flash attention, 64-key chunks, exp2 softmax, maskless fast path
// ---------------------------------------------------------------------------

typedef __attribute__((ext_vector_type(16))) __bf16        v16bf;
typedef __attribute__((ext_vector_type(8)))  float         v8f;
typedef __attribute__((ext_vector_type(8)))  unsigned int  v8ui;

constexpr int S_LEN   = 4096;
constexpr int E_DIM   = 1024;
constexpr int N_HEADS = 16;
constexpr int D_HEAD  = 64;
constexpr int GLOBAL_HEADS = 8;
constexpr int WINDOW  = 256;
constexpr float NEG_INF = -1e30f;
// 1/sqrt(D) * log2(e): scores land in log2 domain -> softmax via exp2f
constexpr float Q_SCALE = 0.125f * 1.44269504088896340736f;

constexpr int LDB = 520;   // padded B-tile row (bf16): 1040B, 16B aligned
constexpr int LDA = 40;    // padded A-tile row: 80B, 8B aligned, conflict-free
constexpr int LDP = 72;    // padded P-tile row: 144B, 16B aligned

// --------------------------- fragment helpers ------------------------------
template <typename P>
__device__ inline v16bf load_a_frag(const P* base, int ld, int lane) {
  const int m  = lane & 15;
  const int hi = (lane >> 4) & 1;
  v8ui r;
#pragma unroll
  for (int j = 0; j < 8; ++j) {
    const int k0 = 2 * (j & 3) + 8 * hi + 16 * (j >> 2);
    r[j] = *reinterpret_cast<const unsigned int*>(base + m * ld + k0);
  }
  return __builtin_bit_cast(v16bf, r);
}

__device__ inline v16bf load_b_frag(const __bf16* base, int ld, int lane) {
  const int n  = lane & 15;
  const int hi = (lane >> 4) & 1;
  v16bf r;
#pragma unroll
  for (int s = 0; s < 16; ++s) {
    const int k = (s < 8) ? (8 * hi + s) : (16 + 8 * hi + (s - 8));
    r[s] = base[(size_t)k * ld + n];
  }
  return r;
}

__device__ inline v8f wmma_bf16(v16bf a, v16bf b, v8f c) {
  return __builtin_amdgcn_wmma_f32_16x16x32_bf16(false, a, false, b,
                                                 (short)0, c, false, false);
}

__device__ inline unsigned pack_bf16(float a, float b) {
  const unsigned short ua = __builtin_bit_cast(unsigned short, (__bf16)a);
  const unsigned short ub = __builtin_bit_cast(unsigned short, (__bf16)b);
  return (unsigned)ua | ((unsigned)ub << 16);
}

__device__ inline float red_max16(float v) {
  v = fmaxf(v, __shfl_xor(v, 1, 32));
  v = fmaxf(v, __shfl_xor(v, 2, 32));
  v = fmaxf(v, __shfl_xor(v, 4, 32));
  v = fmaxf(v, __shfl_xor(v, 8, 32));
  return v;
}
__device__ inline float red_sum16(float v) {
  v += __shfl_xor(v, 1, 32);
  v += __shfl_xor(v, 2, 32);
  v += __shfl_xor(v, 4, 32);
  v += __shfl_xor(v, 8, 32);
  return v;
}

// ---------------------- async global -> LDS helpers ------------------------
// SADDR form: mem = SGPR_base + VGPR_offset; LDS dest = low 32 bits of the
// flat shared address (LDS aperture maps addr[31:0] -> LDS offset).
__device__ inline void async_b128(unsigned lds_off, unsigned voff,
                                  unsigned long long sbase) {
  asm volatile("global_load_async_to_lds_b128 %0, %1, %2"
               :: "v"(lds_off), "v"(voff), "s"(sbase) : "memory");
}
__device__ inline void async_b64(unsigned lds_off, unsigned voff,
                                 unsigned long long sbase) {
  asm volatile("global_load_async_to_lds_b64 %0, %1, %2"
               :: "v"(lds_off), "v"(voff), "s"(sbase) : "memory");
}

// Stage one 32x512 B tile + one 32x32 A tile: exactly 9 async ops per wave,
// so "s_wait_asynccnt 9" at steady state waits for the *previous* tile.
__device__ inline void stage_tiles(__bf16* bufB, __bf16* bufA,
                                   const __bf16* Bsrc, const __bf16* Asrc,
                                   int tid) {
  const unsigned bLds = (unsigned)(size_t)bufB + (tid >> 6) * (LDB * 2) + (tid & 63) * 16;
  const unsigned bGlo = (unsigned)((tid >> 6) * (E_DIM * 2) + (tid & 63) * 16);
  const unsigned long long bBase = (unsigned long long)(size_t)Bsrc;
#pragma unroll
  for (int it = 0; it < 8; ++it)
    async_b128(bLds + it * (4 * LDB * 2), bGlo + it * (4 * E_DIM * 2), bBase);
  const unsigned aLds = (unsigned)(size_t)bufA + (tid >> 3) * (LDA * 2) + (tid & 7) * 8;
  const unsigned aGlo = (unsigned)((tid >> 3) * (E_DIM * 2) + (tid & 7) * 8);
  async_b64(aLds, aGlo, (unsigned long long)(size_t)Asrc);
}

// ---------------------- kernel 0a: x f32 -> bf16 ---------------------------
__global__ __launch_bounds__(256) void xconv_kernel(
    const float* __restrict__ x, __bf16* __restrict__ xb)
{
  const size_t i = ((size_t)blockIdx.x * 256 + threadIdx.x) * 8;
  const float4 a = *reinterpret_cast<const float4*>(x + i);
  const float4 b = *reinterpret_cast<const float4*>(x + i + 4);
  uint4 u;
  u.x = pack_bf16(a.x, a.y);
  u.y = pack_bf16(a.z, a.w);
  u.z = pack_bf16(b.x, b.y);
  u.w = pack_bf16(b.z, b.w);
  *reinterpret_cast<uint4*>(xb + i) = u;
}

// ------------------- kernel 0b: weight transpose + bf16 --------------------
__global__ __launch_bounds__(256) void wtrans_kernel(
    const float* __restrict__ Wq, const float* __restrict__ Wk,
    const float* __restrict__ Wv, const float* __restrict__ Wo,
    __bf16* __restrict__ Wqt, __bf16* __restrict__ Wkt,
    __bf16* __restrict__ Wvt, __bf16* __restrict__ Wot)
{
  __shared__ __bf16 tile[32 * 33];
  const int tid = threadIdx.x;
  const int f0 = blockIdx.x * 32;
  const int e0 = blockIdx.y * 32;
  const int z  = blockIdx.z;
  const float* W = (z == 0) ? Wq : (z == 1) ? Wk : (z == 2) ? Wv : Wo;
  __bf16*     Wt = (z == 0) ? Wqt : (z == 1) ? Wkt : (z == 2) ? Wvt : Wot;

#pragma unroll
  for (int it = 0; it < 4; ++it) {
    const int i = tid + it * 256;
    const int r = i >> 5, c = i & 31;
    tile[c * 33 + r] = (__bf16)W[(size_t)(f0 + r) * E_DIM + e0 + c];
  }
  __syncthreads();
#pragma unroll
  for (int it = 0; it < 4; ++it) {
    const int i = tid + it * 256;
    const int r = i >> 5, c = i & 31;
    Wt[(size_t)(e0 + r) * E_DIM + f0 + c] = tile[r * 33 + c];
  }
}

// --------------------------- kernel 1: QKV proj ----------------------------
// grid (S/32, E/512, 3), block 256 (8 waves); each wave: 32x64 strip,
// 8 wmma per k-step; double-buffered async staging.
__global__ __launch_bounds__(256) void qkv_proj_kernel(
    const __bf16* __restrict__ Xb,
    const __bf16* __restrict__ Wqt, const __bf16* __restrict__ Wkt,
    const __bf16* __restrict__ Wvt,
    __bf16* __restrict__ Qw, __bf16* __restrict__ Ktw, __bf16* __restrict__ Vw)
{
  __shared__ __align__(16) __bf16 ldsA[2][32 * LDA];
  __shared__ __align__(16) __bf16 ldsB[2][32 * LDB];

  const int tid  = threadIdx.x;
  const int lane = tid & 31;
  const int wave = tid >> 5;
  const int s0   = blockIdx.x * 32;
  const int fB   = blockIdx.y * 512;
  const int z    = blockIdx.z;
  const __bf16* Wt = (z == 0) ? Wqt : (z == 1) ? Wkt : Wvt;

  const v8f z8 = {0.f, 0.f, 0.f, 0.f, 0.f, 0.f, 0.f, 0.f};
  v8f acc[2][4] = {{z8, z8, z8, z8}, {z8, z8, z8, z8}};

  stage_tiles(ldsB[0], ldsA[0], Wt + fB, Xb + (size_t)s0 * E_DIM, tid);
  int pb = 0;
  for (int e0 = 0; e0 < E_DIM; e0 += 32, pb ^= 1) {
    const int nxt = e0 + 32;
    if (nxt < E_DIM) {   // prefetch next tile into the other buffer
      stage_tiles(ldsB[pb ^ 1], ldsA[pb ^ 1],
                  Wt + (size_t)nxt * E_DIM + fB,
                  Xb + (size_t)s0 * E_DIM + nxt, tid);
      asm volatile("s_wait_asynccnt 9" ::: "memory");  // previous tile done
    } else {
      asm volatile("s_wait_asynccnt 0" ::: "memory");
    }
    __syncthreads();
    v16bf a[2];
    a[0] = load_a_frag(ldsA[pb],            LDA, lane);
    a[1] = load_a_frag(ldsA[pb] + 16 * LDA, LDA, lane);
#pragma unroll
    for (int t = 0; t < 4; ++t) {
      const v16bf b = load_b_frag(ldsB[pb] + wave * 64 + t * 16, LDB, lane);
      acc[0][t] = wmma_bf16(a[0], b, acc[0][t]);
      acc[1][t] = wmma_bf16(a[1], b, acc[1][t]);
    }
    __syncthreads();
  }

  const int col   = lane & 15;
  const int rhalf = (lane >> 4) * 8;
  if (z == 0) {
#pragma unroll
    for (int mi = 0; mi < 2; ++mi)
#pragma unroll
      for (int t = 0; t < 4; ++t) {
        const int f = fB + wave * 64 + t * 16 + col;
#pragma unroll
        for (int r = 0; r < 8; ++r) {
          const int s = s0 + mi * 16 + rhalf + r;
          Qw[(size_t)s * E_DIM + f] = (__bf16)(acc[mi][t][r] * Q_SCALE);
        }
      }
  } else if (z == 1) {
    // per lane the 8 rows are consecutive s -> pack into one b128 store
#pragma unroll
    for (int mi = 0; mi < 2; ++mi)
#pragma unroll
      for (int t = 0; t < 4; ++t) {
        const int f = fB + wave * 64 + t * 16 + col;
        uint4 u;
        u.x = pack_bf16(acc[mi][t][0], acc[mi][t][1]);
        u.y = pack_bf16(acc[mi][t][2], acc[mi][t][3]);
        u.z = pack_bf16(acc[mi][t][4], acc[mi][t][5]);
        u.w = pack_bf16(acc[mi][t][6], acc[mi][t][7]);
        *reinterpret_cast<uint4*>(Ktw + (size_t)f * S_LEN + s0 + mi * 16 + rhalf) = u;
      }
  } else {
#pragma unroll
    for (int mi = 0; mi < 2; ++mi)
#pragma unroll
      for (int t = 0; t < 4; ++t) {
        const int f = fB + wave * 64 + t * 16 + col;
#pragma unroll
        for (int r = 0; r < 8; ++r) {
          const int s = s0 + mi * 16 + rhalf + r;
          Vw[(size_t)s * E_DIM + f] = (__bf16)acc[mi][t][r];
        }
      }
  }
}

// --------------------------- kernel 2: attention ---------------------------
__global__ __launch_bounds__(32) void attn_kernel(
    const __bf16* __restrict__ Qw, const __bf16* __restrict__ Ktw,
    const __bf16* __restrict__ Vw, __bf16* __restrict__ Aw)
{
  __shared__ __align__(16) __bf16 ldsP[16 * LDP];

  const int lane = threadIdx.x & 31;
  const int s0   = blockIdx.x * 16;
  const int h    = blockIdx.y;
  const bool is_local = (h >= GLOBAL_HEADS);

  const v16bf qa0 = load_a_frag(Qw + (size_t)s0 * E_DIM + h * D_HEAD,      E_DIM, lane);
  const v16bf qa1 = load_a_frag(Qw + (size_t)s0 * E_DIM + h * D_HEAD + 32, E_DIM, lane);

  const v8f z8 = {0.f, 0.f, 0.f, 0.f, 0.f, 0.f, 0.f, 0.f};
  v8f o[4] = {z8, z8, z8, z8};
  float rmax[8], rsum[8];
#pragma unroll
  for (int r = 0; r < 8; ++r) { rmax[r] = -3e38f; rsum[r] = 0.f; }

  const int col   = lane & 15;
  const int rhalf = (lane >> 4) * 8;

  const int cEnd = (s0 + 15) >> 6;
  int cBeg = 0;
  if (is_local) { const int lo = s0 - WINDOW; if (lo > 0) cBeg = lo >> 6; }

  for (int c = cBeg; c <= cEnd; ++c) {
    const int tb = c * 64;

    v8f sct[4];
#pragma unroll
    for (int tt = 0; tt < 4; ++tt) {
      v16bf kb = load_b_frag(Ktw + (size_t)(h * D_HEAD) * S_LEN + tb + tt * 16, S_LEN, lane);
      sct[tt] = wmma_bf16(qa0, kb, z8);
      kb = load_b_frag(Ktw + (size_t)(h * D_HEAD + 32) * S_LEN + tb + tt * 16, S_LEN, lane);
      sct[tt] = wmma_bf16(qa1, kb, sct[tt]);
    }

    const bool need_mask =
        (tb + 63 > s0) || (is_local && (tb < s0 + 15 - WINDOW));
    float scale_r[8];
    if (need_mask) {
#pragma unroll
      for (int r = 0; r < 8; ++r) {
        const int s   = s0 + rhalf + r;
        const int row = rhalf + r;
        float v[4];
#pragma unroll
        for (int tt = 0; tt < 4; ++tt) {
          const int t = tb + tt * 16 + col;
          const bool ok = (t <= s) && (!is_local || t >= s - WINDOW);
          v[tt] = ok ? sct[tt][r] : NEG_INF;
        }
        float mx = red_max16(fmaxf(fmaxf(v[0], v[1]), fmaxf(v[2], v[3])));
        const float nm  = fmaxf(rmax[r], mx);
        const float scl = exp2f(rmax[r] - nm);
        float rs = 0.f;
#pragma unroll
        for (int tt = 0; tt < 4; ++tt) {
          const float pp = (v[tt] < -1e29f) ? 0.f : exp2f(v[tt] - nm);
          rs += pp;
          ldsP[row * LDP + tt * 16 + col] = (__bf16)pp;
        }
        rsum[r] = rsum[r] * scl + red_sum16(rs);
        rmax[r] = nm;
        scale_r[r] = scl;
      }
    } else {  // fully-valid interior chunk
#pragma unroll
      for (int r = 0; r < 8; ++r) {
        const int row = rhalf + r;
        const float v0 = sct[0][r], v1 = sct[1][r];
        const float v2 = sct[2][r], v3 = sct[3][r];
        float mx = red_max16(fmaxf(fmaxf(v0, v1), fmaxf(v2, v3)));
        const float nm  = fmaxf(rmax[r], mx);
        const float scl = exp2f(rmax[r] - nm);
        const float p0 = exp2f(v0 - nm), p1 = exp2f(v1 - nm);
        const float p2 = exp2f(v2 - nm), p3 = exp2f(v3 - nm);
        ldsP[row * LDP + col]      = (__bf16)p0;
        ldsP[row * LDP + 16 + col] = (__bf16)p1;
        ldsP[row * LDP + 32 + col] = (__bf16)p2;
        ldsP[row * LDP + 48 + col] = (__bf16)p3;
        rsum[r] = rsum[r] * scl + red_sum16(p0 + p1 + p2 + p3);
        rmax[r] = nm;
        scale_r[r] = scl;
      }
    }
    asm volatile("s_wait_dscnt 0" ::: "memory");

#pragma unroll
    for (int dd = 0; dd < 4; ++dd)
#pragma unroll
      for (int r = 0; r < 8; ++r) o[dd][r] *= scale_r[r];

    const v16bf pa0 = load_a_frag(ldsP,      LDP, lane);
    const v16bf pa1 = load_a_frag(ldsP + 32, LDP, lane);
#pragma unroll
    for (int dd = 0; dd < 4; ++dd) {
      v16bf vb = load_b_frag(Vw + (size_t)tb * E_DIM + h * D_HEAD + dd * 16, E_DIM, lane);
      o[dd] = wmma_bf16(pa0, vb, o[dd]);
      vb = load_b_frag(Vw + (size_t)(tb + 32) * E_DIM + h * D_HEAD + dd * 16, E_DIM, lane);
      o[dd] = wmma_bf16(pa1, vb, o[dd]);
    }
  }

#pragma unroll
  for (int r = 0; r < 8; ++r) {
    const int s = s0 + rhalf + r;
    const float inv = 1.f / rsum[r];
    __bf16* dst = Aw + (size_t)s * E_DIM + h * D_HEAD + col;
#pragma unroll
    for (int dd = 0; dd < 4; ++dd) dst[dd * 16] = (__bf16)(o[dd][r] * inv);
  }
}

// --------------------------- kernel 3: output proj -------------------------
__global__ __launch_bounds__(256) void out_proj_kernel(
    const __bf16* __restrict__ Aw, const __bf16* __restrict__ Wot,
    const float* __restrict__ bo, float* __restrict__ out)
{
  __shared__ __align__(16) __bf16 ldsA[2][32 * LDA];
  __shared__ __align__(16) __bf16 ldsB[2][32 * LDB];

  const int tid  = threadIdx.x;
  const int lane = tid & 31;
  const int wave = tid >> 5;
  const int s0   = blockIdx.x * 32;
  const int fB   = blockIdx.y * 512;

  const v8f z8 = {0.f, 0.f, 0.f, 0.f, 0.f, 0.f, 0.f, 0.f};
  v8f acc[2][4] = {{z8, z8, z8, z8}, {z8, z8, z8, z8}};

  stage_tiles(ldsB[0], ldsA[0], Wot + fB, Aw + (size_t)s0 * E_DIM, tid);
  int pb = 0;
  for (int e0 = 0; e0 < E_DIM; e0 += 32, pb ^= 1) {
    const int nxt = e0 + 32;
    if (nxt < E_DIM) {
      stage_tiles(ldsB[pb ^ 1], ldsA[pb ^ 1],
                  Wot + (size_t)nxt * E_DIM + fB,
                  Aw + (size_t)s0 * E_DIM + nxt, tid);
      asm volatile("s_wait_asynccnt 9" ::: "memory");
    } else {
      asm volatile("s_wait_asynccnt 0" ::: "memory");
    }
    __syncthreads();
    v16bf a[2];
    a[0] = load_a_frag(ldsA[pb],            LDA, lane);
    a[1] = load_a_frag(ldsA[pb] + 16 * LDA, LDA, lane);
#pragma unroll
    for (int t = 0; t < 4; ++t) {
      const v16bf b = load_b_frag(ldsB[pb] + wave * 64 + t * 16, LDB, lane);
      acc[0][t] = wmma_bf16(a[0], b, acc[0][t]);
      acc[1][t] = wmma_bf16(a[1], b, acc[1][t]);
    }
    __syncthreads();
  }

  const int col   = lane & 15;
  const int rhalf = (lane >> 4) * 8;
#pragma unroll
  for (int mi = 0; mi < 2; ++mi)
#pragma unroll
    for (int t = 0; t < 4; ++t) {
      const int f = fB + wave * 64 + t * 16 + col;
      const float bias = bo[f];
#pragma unroll
      for (int r = 0; r < 8; ++r) {
        const int s = s0 + mi * 16 + rhalf + r;
        out[(size_t)s * E_DIM + f] = acc[mi][t][r] + bias;
      }
    }
}

// ------------------------------- launcher ----------------------------------
extern "C" void kernel_launch(void* const* d_in, const int* in_sizes, int n_in,
                              void* d_out, int out_size, void* d_ws, size_t ws_size,
                              hipStream_t stream) {
  const float* x  = (const float*)d_in[0];
  const float* Wq = (const float*)d_in[1];
  const float* Wk = (const float*)d_in[2];
  const float* Wv = (const float*)d_in[3];
  const float* Wo = (const float*)d_in[4];
  const float* bo = (const float*)d_in[5];
  float* out = (float*)d_out;

  const size_t SE = (size_t)S_LEN * E_DIM;
  const size_t EE = (size_t)E_DIM * E_DIM;
  __bf16* Qw  = (__bf16*)d_ws;     // [S][E]  (pre-scaled by 1/8 * log2 e)
  __bf16* Ktw = Qw  + SE;          // [E][S]  (K transposed)
  __bf16* Vw  = Ktw + SE;          // [S][E]
  __bf16* Aw  = Vw  + SE;          // [S][E]  attention output
  __bf16* Wqt = Aw  + SE;          // [E][E]  transposed bf16 weights
  __bf16* Wkt = Wqt + EE;
  __bf16* Wvt = Wkt + EE;
  __bf16* Wot = Wvt + EE;
  __bf16* Xb  = Wot + EE;          // [S][E]  bf16 activations

  xconv_kernel<<<dim3((unsigned)(SE / (256 * 8))), 256, 0, stream>>>(x, Xb);
  wtrans_kernel<<<dim3(E_DIM / 32, E_DIM / 32, 4), 256, 0, stream>>>(
      Wq, Wk, Wv, Wo, Wqt, Wkt, Wvt, Wot);
  qkv_proj_kernel<<<dim3(S_LEN / 32, E_DIM / 512, 3), 256, 0, stream>>>(
      Xb, Wqt, Wkt, Wvt, Qw, Ktw, Vw);
  attn_kernel<<<dim3(S_LEN / 16, N_HEADS), 32, 0, stream>>>(Qw, Ktw, Vw, Aw);
  out_proj_kernel<<<dim3(S_LEN / 32, E_DIM / 512), 256, 0, stream>>>(
      Aw, Wot, bo, out);
}
